// TinyAttn_87883620811147
// MI455X (gfx1250) — compile-verified, matrix operand order checked by reading, not run on hardware
//
#include <hip/hip_runtime.h>

// ---------------------------------------------------------------------------
// TinyAttn forward for gfx1250 (MI455X, wave32, WMMA).
//   y = (softmax( rope(xWq^T) rope(xWk^T)^T / 8 ) (xWv^T)) Wo^T
// Stages:
//   0) f32 -> bf16 pre-convert of x and Wq/Wk/Wv/Wo (bandwidth-bound, once)
//   1) QKV GEMM (bf16 WMMA, f32 acc) + fused RoPE; V stored transposed/head
//   2) flash attention, all operands bf16, contiguous-pair fragment loads
//   3) output projection GEMM -> f32
// All matrix math via v_wmma_f32_16x16x32_bf16.
// ---------------------------------------------------------------------------

#define DIMD   1024
#define SEQT   2048
#define BATCH  4
#define HEADS  16
#define HDIM   64

typedef __attribute__((ext_vector_type(16))) __bf16 v16bf;
typedef __attribute__((ext_vector_type(8)))  float  v8f;

// ---- WMMA wrapper ----------------------------------------------------------
__device__ __forceinline__ v8f wmma_bf16(v16bf a, v16bf b, v8f c) {
  // (neg_a, A, neg_b, B, c_mod, C, reuse_a, reuse_b)
  return __builtin_amdgcn_wmma_f32_16x16x32_bf16(false, a, false, b,
                                                 (short)0, c, false, false);
}

// K index covered by fragment VGPR i (0..7), per ISA 16-bit A/B layout:
//   VGPR i packs K = kb + ((i&4)<<2) + ((i&3)<<1)  and K+1
//   kb = 0 for lanes 0..15, 8 for lanes 16..31
__device__ __forceinline__ int frag_k(int i, int lane) {
  return ((lane & 16) ? 8 : 0) + ((i & 4) << 2) + ((i & 3) << 1);
}

// A or B fragment from a row-major bf16 matrix where the contraction (K)
// dimension is contiguous in memory.  Covers:
//   A = x / Q / P tiles              (row = M)
//   B = W tiles for x*W^T            (row = N, B[k][n] = W[n][k])
//   B = K tiles for Q*K^T            (row = key)
//   B = Vt tiles for P*V             (row = d,  Vt[d][t])
// Pairs are contiguous -> compiler merges into global_load_b128.
__device__ __forceinline__ v16bf frag_rm_bf16(const __bf16* __restrict__ p0,
                                              int ld, int lane) {
  const int r = lane & 15;
  v16bf f;
#pragma unroll
  for (int i = 0; i < 8; ++i) {
    const int k = frag_k(i, lane);
    const __bf16* p = p0 + (size_t)r * ld + k;
    f[2 * i]     = p[0];
    f[2 * i + 1] = p[1];
  }
  return f;
}

// ---------------------------------------------------------------------------
// Kernel 0: f32 -> bf16 convert (grid-stride, float4-wide).
// ---------------------------------------------------------------------------
__global__ __launch_bounds__(256)
void cvt_bf16_kernel(const float* __restrict__ src, __bf16* __restrict__ dst,
                     unsigned n) {
  unsigned i = (blockIdx.x * blockDim.x + threadIdx.x) * 4u;
  const unsigned stride = gridDim.x * blockDim.x * 4u;
  for (; i < n; i += stride) {
    const float4 v = *(const float4*)(src + i);
    dst[i + 0] = (__bf16)v.x;
    dst[i + 1] = (__bf16)v.y;
    dst[i + 2] = (__bf16)v.z;
    dst[i + 3] = (__bf16)v.w;
  }
}

// ---------------------------------------------------------------------------
// Kernel 1: Q/K/V = x @ W^T (bf16 WMMA), fused RoPE on Q and K.
// Wave tile: 32 rows x 64 cols (one head wide).  blockIdx.z picks Wq/Wk/Wv.
// V is stored transposed per head: Vt[b][h][d][t].
// ---------------------------------------------------------------------------
__global__ __launch_bounds__(128)
void qkv_rope_kernel(const __bf16* __restrict__ xb,
                     const __bf16* __restrict__ Wqb,
                     const __bf16* __restrict__ Wkb,
                     const __bf16* __restrict__ Wvb,
                     __bf16* __restrict__ Qo,
                     __bf16* __restrict__ Ko,
                     __bf16* __restrict__ Vt) {
  const int lane = threadIdx.x & 31;
  const int wave = threadIdx.x >> 5;
  const int rowBase = (blockIdx.x * 4 + wave) * 32;   // row in flattened (B*T)
  const int n0 = blockIdx.y * 64;                     // output column base
  const int which = blockIdx.z;

  const __bf16* W = (which == 0) ? Wqb : (which == 1) ? Wkb : Wvb;

  v8f acc[2][4];
#pragma unroll
  for (int r0 = 0; r0 < 2; ++r0)
#pragma unroll
    for (int j = 0; j < 4; ++j)
#pragma unroll
      for (int e = 0; e < 8; ++e) acc[r0][j][e] = 0.0f;

  for (int k0 = 0; k0 < DIMD; k0 += 32) {
    const v16bf a0 = frag_rm_bf16(xb + (size_t)rowBase * DIMD + k0,        DIMD, lane);
    const v16bf a1 = frag_rm_bf16(xb + (size_t)(rowBase + 16) * DIMD + k0, DIMD, lane);
#pragma unroll
    for (int j = 0; j < 4; ++j) {
      const v16bf b = frag_rm_bf16(W + (size_t)(n0 + j * 16) * DIMD + k0, DIMD, lane);
      acc[0][j] = wmma_bf16(a0, b, acc[0][j]);
      acc[1][j] = wmma_bf16(a1, b, acc[1][j]);
    }
  }

  // Epilogue: C layout -> lane holds col n = lane&15, rows M = mh + i.
  const int n  = lane & 15;
  const int mh = (lane & 16) ? 8 : 0;
  const int h  = blockIdx.y;                 // 64-wide tile == one head
  // inv_freq = 10000^(-n/16);  ln(10000)/16 = 0.57564627...
  const float invf = __expf(-(float)n * 0.57564627324851142f);

#pragma unroll
  for (int r0 = 0; r0 < 2; ++r0) {
#pragma unroll
    for (int i = 0; i < 8; ++i) {
      const int r = rowBase + r0 * 16 + mh + i;   // flattened b*T + t
      const int t = r & (SEQT - 1);
      if (which < 2) {
        // RoPE: subtile 0 -> d = n (<16), subtile 1 -> d = 16+n
        const float ang = (float)t * invf;
        const float cs = __cosf(ang), sn = __sinf(ang);
        const float c0 = acc[r0][0][i], c1 = acc[r0][1][i];
        acc[r0][0][i] = c0 * cs + c1 * sn;
        acc[r0][1][i] = c1 * cs - c0 * sn;
        __bf16* Out = (which == 0) ? Qo : Ko;
#pragma unroll
        for (int j = 0; j < 4; ++j)
          Out[(size_t)r * DIMD + n0 + j * 16 + n] = (__bf16)acc[r0][j][i];
      } else {
        // V transposed: Vt[((b*H + h)*HDIM + d) * T + t]
        const int b = r >> 11;                    // r / SEQT
#pragma unroll
        for (int j = 0; j < 4; ++j) {
          const int d = j * 16 + n;
          Vt[(((size_t)b * HEADS + h) * HDIM + d) * SEQT + t] =
              (__bf16)acc[r0][j][i];
        }
      }
    }
  }
}

// ---------------------------------------------------------------------------
// Kernel 2: flash attention (non-causal).  One wave = 16 queries of one
// (batch, head); loops over keys in blocks of 32.  P round-trips through LDS
// to convert the f32 C-layout into the bf16 A-layout for the P@V WMMAs.
// V comes from the transposed Vt buffer so every fragment load is a
// contiguous-pair (b128-vectorizable) access.
// ---------------------------------------------------------------------------
__global__ __launch_bounds__(128)
void flash_attn_kernel(const __bf16* __restrict__ Q,
                       const __bf16* __restrict__ K,
                       const __bf16* __restrict__ Vt,
                       __bf16* __restrict__ O) {
  __shared__ __bf16 Pbuf[4][16][32];

  const int lane = threadIdx.x & 31;
  const int wave = threadIdx.x >> 5;
  const int qBase = blockIdx.x * 64 + wave * 16;
  const int h = blockIdx.y;
  const int b = blockIdx.z;

  const size_t base = ((size_t)b * SEQT) * DIMD + (size_t)h * HDIM;
  const __bf16* Qb  = Q + base;
  const __bf16* Kb  = K + base;
  const __bf16* Vtb = Vt + ((size_t)b * HEADS + h) * HDIM * SEQT;

  // Q fragments are loop-invariant: 16 queries x 64 head dims = 2 A-frags.
  const v16bf qa0 = frag_rm_bf16(Qb + (size_t)qBase * DIMD + 0,  DIMD, lane);
  const v16bf qa1 = frag_rm_bf16(Qb + (size_t)qBase * DIMD + 32, DIMD, lane);

  v8f o[4];
#pragma unroll
  for (int j = 0; j < 4; ++j)
#pragma unroll
    for (int e = 0; e < 8; ++e) o[j][e] = 0.0f;

  float mrow[8], lrow[8];
#pragma unroll
  for (int i = 0; i < 8; ++i) { mrow[i] = -1.0e30f; lrow[i] = 0.0f; }

  const int n  = lane & 15;
  const int mh = (lane & 16) ? 8 : 0;
  const float scale = 0.125f;   // 1/sqrt(64)

  for (int kb = 0; kb < SEQT; kb += 32) {
    if (kb + 32 < SEQT) {
      __builtin_prefetch(Kb + (size_t)(kb + 32) * DIMD, 0, 1);
      __builtin_prefetch(Vtb + (size_t)(kb + 32), 0, 1);
    }

    // S = Q K^T for this 16q x 32k block (contraction over 64 head dims)
    v8f s[2];
#pragma unroll
    for (int j = 0; j < 2; ++j) {
#pragma unroll
      for (int e = 0; e < 8; ++e) s[j][e] = 0.0f;
      const v16bf b0 = frag_rm_bf16(Kb + (size_t)(kb + j * 16) * DIMD + 0,  DIMD, lane);
      s[j] = wmma_bf16(qa0, b0, s[j]);
      const v16bf b1 = frag_rm_bf16(Kb + (size_t)(kb + j * 16) * DIMD + 32, DIMD, lane);
      s[j] = wmma_bf16(qa1, b1, s[j]);
    }

    // Online softmax.  Row M = mh+i lives in one VGPR across 16 lanes of a
    // half-wave -> reduce with shfl_xor {1,2,4,8}.
#pragma unroll
    for (int i = 0; i < 8; ++i) {
      const float s0 = s[0][i] * scale;
      const float s1 = s[1][i] * scale;
      float v = fmaxf(s0, s1);
      v = fmaxf(v, __shfl_xor(v, 1, 32));
      v = fmaxf(v, __shfl_xor(v, 2, 32));
      v = fmaxf(v, __shfl_xor(v, 4, 32));
      v = fmaxf(v, __shfl_xor(v, 8, 32));
      const float mnew = fmaxf(mrow[i], v);
      const float corr = __expf(mrow[i] - mnew);
      const float p0 = __expf(s0 - mnew);
      const float p1 = __expf(s1 - mnew);
      float ps = p0 + p1;
      ps += __shfl_xor(ps, 1, 32);
      ps += __shfl_xor(ps, 2, 32);
      ps += __shfl_xor(ps, 4, 32);
      ps += __shfl_xor(ps, 8, 32);
      lrow[i] = lrow[i] * corr + ps;
      mrow[i] = mnew;
#pragma unroll
      for (int j = 0; j < 4; ++j) o[j][i] *= corr;
      Pbuf[wave][mh + i][n]      = (__bf16)p0;   // key = n
      Pbuf[wave][mh + i][16 + n] = (__bf16)p1;   // key = 16+n
    }
    __syncthreads();

    // O += P @ V  (P: 16x32 A-frag from LDS; B[k][n] = Vt[d = j*16+n][kb+k])
    const v16bf pa = frag_rm_bf16(&Pbuf[wave][0][0], 32, lane);
#pragma unroll
    for (int j = 0; j < 4; ++j) {
      const v16bf vb = frag_rm_bf16(Vtb + (size_t)(j * 16) * SEQT + kb, SEQT, lane);
      o[j] = wmma_bf16(pa, vb, o[j]);
    }
    __syncthreads();
  }

  // Epilogue: divide by l and store bf16 (row-major B*T x D).
#pragma unroll
  for (int i = 0; i < 8; ++i) {
    const float inv = 1.0f / lrow[i];
    const size_t r = (size_t)(qBase + mh + i);
#pragma unroll
    for (int j = 0; j < 4; ++j)
      O[base + r * DIMD + j * 16 + n] = (__bf16)(o[j][i] * inv);
  }
}

// ---------------------------------------------------------------------------
// Kernel 3: y = AO @ Wo^T, f32 output.  32x64 wave tile like kernel 1.
// ---------------------------------------------------------------------------
__global__ __launch_bounds__(128)
void out_proj_kernel(const __bf16* __restrict__ AO,
                     const __bf16* __restrict__ Wob,
                     float* __restrict__ Y) {
  const int lane = threadIdx.x & 31;
  const int wave = threadIdx.x >> 5;
  const int rowBase = (blockIdx.x * 4 + wave) * 32;
  const int n0 = blockIdx.y * 64;

  v8f acc[2][4];
#pragma unroll
  for (int r0 = 0; r0 < 2; ++r0)
#pragma unroll
    for (int j = 0; j < 4; ++j)
#pragma unroll
      for (int e = 0; e < 8; ++e) acc[r0][j][e] = 0.0f;

  for (int k0 = 0; k0 < DIMD; k0 += 32) {
    const v16bf a0 = frag_rm_bf16(AO + (size_t)rowBase * DIMD + k0,        DIMD, lane);
    const v16bf a1 = frag_rm_bf16(AO + (size_t)(rowBase + 16) * DIMD + k0, DIMD, lane);
#pragma unroll
    for (int j = 0; j < 4; ++j) {
      const v16bf b = frag_rm_bf16(Wob + (size_t)(n0 + j * 16) * DIMD + k0, DIMD, lane);
      acc[0][j] = wmma_bf16(a0, b, acc[0][j]);
      acc[1][j] = wmma_bf16(a1, b, acc[1][j]);
    }
  }

  const int n  = lane & 15;
  const int mh = (lane & 16) ? 8 : 0;
#pragma unroll
  for (int r0 = 0; r0 < 2; ++r0)
#pragma unroll
    for (int i = 0; i < 8; ++i) {
      const size_t r = (size_t)(rowBase + r0 * 16 + mh + i);
#pragma unroll
      for (int j = 0; j < 4; ++j)
        Y[r * DIMD + n0 + j * 16 + n] = acc[r0][j][i];
    }
}

// ---------------------------------------------------------------------------
// Launcher.  Inputs: x, Wq, Wk, Wv, Wo (all f32).  Output: y (f32).
// Workspace (bf16): xb(16M) | Wq|Wk|Wv|Wo (2M each) | Q(16M) | K(16M) |
//                   Vt(16M) | AO(16M)  => 88 MB total.
// ---------------------------------------------------------------------------
extern "C" void kernel_launch(void* const* d_in, const int* in_sizes, int n_in,
                              void* d_out, int out_size, void* d_ws, size_t ws_size,
                              hipStream_t stream) {
  const float* x  = (const float*)d_in[0];
  const float* Wq = (const float*)d_in[1];
  const float* Wk = (const float*)d_in[2];
  const float* Wv = (const float*)d_in[3];
  const float* Wo = (const float*)d_in[4];
  float* y = (float*)d_out;

  const size_t NBTD = (size_t)BATCH * SEQT * DIMD;   // 8,388,608 elems
  const size_t NW   = (size_t)DIMD * DIMD;           // 1,048,576 elems

  __bf16* xb  = (__bf16*)d_ws;
  __bf16* Wqb = xb + NBTD;
  __bf16* Wkb = Wqb + NW;
  __bf16* Wvb = Wkb + NW;
  __bf16* Wob = Wvb + NW;
  __bf16* Qw  = Wob + NW;
  __bf16* Kw  = Qw + NBTD;
  __bf16* Vtw = Kw + NBTD;
  __bf16* AOw = Vtw + NBTD;

  // Stage 0: f32 -> bf16 pre-conversion (bandwidth-bound, ~70 MB traffic)
  cvt_bf16_kernel<<<dim3((unsigned)(NBTD / 4 / 256)), dim3(256), 0, stream>>>(x,  xb,  (unsigned)NBTD);
  cvt_bf16_kernel<<<dim3((unsigned)(NW   / 4 / 256)), dim3(256), 0, stream>>>(Wq, Wqb, (unsigned)NW);
  cvt_bf16_kernel<<<dim3((unsigned)(NW   / 4 / 256)), dim3(256), 0, stream>>>(Wk, Wkb, (unsigned)NW);
  cvt_bf16_kernel<<<dim3((unsigned)(NW   / 4 / 256)), dim3(256), 0, stream>>>(Wv, Wvb, (unsigned)NW);
  cvt_bf16_kernel<<<dim3((unsigned)(NW   / 4 / 256)), dim3(256), 0, stream>>>(Wo, Wob, (unsigned)NW);

  const int Mrows   = BATCH * SEQT;          // 8192
  const int Mtiles  = Mrows / 32;            // 256 (32-row wave tiles)
  const int Ntiles  = DIMD / 64;             // 16

  dim3 blk(128);
  dim3 g1(Mtiles / 4, Ntiles, 3);
  qkv_rope_kernel<<<g1, blk, 0, stream>>>(xb, Wqb, Wkb, Wvb, Qw, Kw, Vtw);

  dim3 g2(SEQT / 64, HEADS, BATCH);
  flash_attn_kernel<<<g2, blk, 0, stream>>>(Qw, Kw, Vtw, AOw);

  dim3 g3(Mtiles / 4, Ntiles, 1);
  out_proj_kernel<<<g3, blk, 0, stream>>>(AOw, Wob, y);
}
